// GlobalAttentionTransformer_31379031064666
// MI455X (gfx1250) — matrix operand order repaired and checked by reading, not compiled, not run on hardware
//
#include <hip/hip_runtime.h>
#include <hip/hip_bf16.h>
#include <math.h>

typedef __bf16 bf16;
typedef __attribute__((ext_vector_type(16))) __bf16 v16bf;
typedef __attribute__((ext_vector_type(8)))  __bf16 v8bf;
typedef __attribute__((ext_vector_type(8)))  float  v8f;

#define DMODEL 768
#define NHEADS 12
#define DHEAD  64
#define BATCH  8
#define NPIX   1024
#define GTOK   5
#define SEQ    1029          /* GTOK + NPIX */
#define SPAD   1056          /* 66*16, divisible by 32 */
#define MTOT   (BATCH*SPAD)  /* 8448 */
#define D3     2304
#define DFF    3072

static const size_t CTX_OFF = (size_t)BATCH*DMODEL*NPIX;              /* 6291456 */
static const size_t REG_OFF = CTX_OFF + (size_t)BATCH*DMODEL;         /* 6297600 */

union FragU { v16bf v; v8bf h[2]; };

/* ---- WMMA fragment loaders from global (CDNA5 16x16x32 bf16 VGPR layouts) ---- */
__device__ __forceinline__ v16bf load_frag_a(const bf16* base, int stride, int lane) {
  int half = lane >> 4, r = lane & 15;
  const bf16* p = base + (size_t)r * stride + half * 8;
  FragU f;
  f.h[0] = *(const v8bf*)(p);        /* K 0..7  (lanes<16)  / 8..15  */
  f.h[1] = *(const v8bf*)(p + 16);   /* K 16..23 (lanes<16) / 24..31 */
  return f.v;
}
__device__ __forceinline__ v16bf load_frag_b(const bf16* base, int stride, int lane) {
  int half = lane >> 4, n = lane & 15;
  const bf16* p = base + (size_t)n * stride + half * 8; /* lanes<16: K0..7,16..23; lanes>=16: K8..15,24..31 */
  FragU f;
  f.h[0] = *(const v8bf*)(p);
  f.h[1] = *(const v8bf*)(p + 16);
  return f.v;
}
/* ---- fragment loaders from LDS staging tiles ---- */
__device__ __forceinline__ v16bf frag_a_lds(const bf16 (&buf)[64][32], int row0, int lane) {
  int half = lane >> 4, r = row0 + (lane & 15);
  FragU f;
#pragma unroll
  for (int i = 0; i < 8; ++i) {
    f.h[0][i] = buf[r][half * 8 + i];
    f.h[1][i] = buf[r][half * 8 + 16 + i];
  }
  return f.v;
}
__device__ __forceinline__ v16bf frag_b_lds(const bf16 (&buf)[128][32], int row0, int lane) {
  int half = lane >> 4, n = row0 + (lane & 15);
  FragU f;
#pragma unroll
  for (int i = 0; i < 8; ++i) {
    f.h[0][i] = buf[n][half * 8 + i];
    f.h[1][i] = buf[n][half * 8 + 16 + i];
  }
  return f.v;
}
__device__ __forceinline__ v8f wmma_bf(v16bf a, v16bf b, v8f c) {
  return __builtin_amdgcn_wmma_f32_16x16x32_bf16(false, a, false, b, (short)0, c, false, false);
}

/* ---- CDNA5 async global->LDS (16B per lane), tracked by ASYNCcnt ---- */
__device__ __forceinline__ void async_load16(unsigned lds_off, const bf16* gaddr) {
  asm volatile("global_load_async_to_lds_b128 %0, %1, off"
               :: "v"(lds_off), "v"(gaddr) : "memory");
}
__device__ __forceinline__ void wait_asynccnt0() {
  asm volatile("s_wait_asynccnt 0x0" ::: "memory");
}

/* ---- weight f32 -> bf16 ---- */
__global__ void cvt_bf16_kernel(const float* __restrict__ src, bf16* __restrict__ dst, int n) {
  int i = blockIdx.x * 256 + threadIdx.x;
  if (i < n) dst[i] = (bf16)src[i];
}

/* ---- token assembly + positional encoding + LayerNorm1 ---- */
__global__ void prep_ln1_kernel(const float* __restrict__ x, const float* __restrict__ ctx,
                                const float* __restrict__ reg, const float* __restrict__ lns,
                                const float* __restrict__ lnb,
                                float* __restrict__ xwt, bf16* __restrict__ xwt_bf) {
  int row = blockIdx.x;
  int b = row / SPAD, s = row - b * SPAD;
  int tid = threadIdx.x;
  __shared__ float red[256];
  if (s >= SEQ) {   /* zero padding rows */
    for (int j = 0; j < 3; ++j) {
      int ch = tid + 256 * j;
      xwt[(size_t)row * DMODEL + ch] = 0.0f;
      xwt_bf[(size_t)row * DMODEL + ch] = (bf16)0.0f;
    }
    return;
  }
  float v[3];
  for (int j = 0; j < 3; ++j) {
    int ch = tid + 256 * j;
    float val;
    if (s < GTOK) {
      float tok = (s == 0) ? ctx[ch] : reg[(s - 1) * DMODEL + ch];
      float ex = (float)(2 * (ch >> 1)) / (float)DMODEL;
      float angle = (float)s / powf(10000.0f, ex);
      val = tok + ((ch & 1) ? cosf(angle) : sinf(angle));
    } else {
      val = x[(size_t)b * DMODEL * NPIX + (size_t)ch * NPIX + (s - GTOK)];
    }
    v[j] = val;
  }
  float lsum = v[0] + v[1] + v[2];
  red[tid] = lsum; __syncthreads();
  for (int off = 128; off > 0; off >>= 1) { if (tid < off) red[tid] += red[tid + off]; __syncthreads(); }
  float mean = red[0] / (float)DMODEL; __syncthreads();
  float lvar = 0.0f;
  for (int j = 0; j < 3; ++j) { float d = v[j] - mean; lvar += d * d; }
  red[tid] = lvar; __syncthreads();
  for (int off = 128; off > 0; off >>= 1) { if (tid < off) red[tid] += red[tid + off]; __syncthreads(); }
  float inv = rsqrtf(red[0] / (float)DMODEL + 1e-5f);
  for (int j = 0; j < 3; ++j) {
    int ch = tid + 256 * j;
    float xn = (v[j] - mean) * inv * lns[ch] + lnb[ch];
    xwt[(size_t)row * DMODEL + ch] = xn;
    xwt_bf[(size_t)row * DMODEL + ch] = (bf16)xn;
  }
}

/* ---- LayerNorm2 (y -> bf16) ---- */
__global__ void ln2_kernel(const float* __restrict__ y, const float* __restrict__ lns,
                           const float* __restrict__ lnb, bf16* __restrict__ dst) {
  int row = blockIdx.x;
  int s = row % SPAD;
  int tid = threadIdx.x;
  __shared__ float red[256];
  if (s >= SEQ) {
    for (int j = 0; j < 3; ++j) dst[(size_t)row * DMODEL + tid + 256 * j] = (bf16)0.0f;
    return;
  }
  float v[3];
  for (int j = 0; j < 3; ++j) v[j] = y[(size_t)row * DMODEL + tid + 256 * j];
  float lsum = v[0] + v[1] + v[2];
  red[tid] = lsum; __syncthreads();
  for (int off = 128; off > 0; off >>= 1) { if (tid < off) red[tid] += red[tid + off]; __syncthreads(); }
  float mean = red[0] / (float)DMODEL; __syncthreads();
  float lvar = 0.0f;
  for (int j = 0; j < 3; ++j) { float d = v[j] - mean; lvar += d * d; }
  red[tid] = lvar; __syncthreads();
  for (int off = 128; off > 0; off >>= 1) { if (tid < off) red[tid] += red[tid + off]; __syncthreads(); }
  float inv = rsqrtf(red[0] / (float)DMODEL + 1e-5f);
  for (int j = 0; j < 3; ++j) {
    int ch = tid + 256 * j;
    dst[(size_t)row * DMODEL + ch] = (bf16)((v[j] - mean) * inv * lns[ch] + lnb[ch]);
  }
}

/* ---- generic WMMA GEMM: out = A(MxK) @ W(NxK)^T ----
   128 threads (4 waves, 2x2), block tile 64x128, K-step 32.
   Async double-buffered LDS staging: tile t+1 streams in via
   global_load_async_to_lds_b128 while tile t feeds 8 WMMAs/wave.
   MODE 0: +bias, scatter to q/k/vT (bf16)
   MODE 1: +bias + resid(xwt) -> y (f32)
   MODE 2: +bias, GELU -> out_bf
   MODE 3: +bias + resid(y) -> scatter to d_out (feature/context/register) */
template<int MODE>
__global__ void gemm_wmma_kernel(const bf16* __restrict__ A, const bf16* __restrict__ W,
                                 const float* __restrict__ bias, int K, int N,
                                 const float* __restrict__ resid, float* __restrict__ out_f,
                                 bf16* __restrict__ out_bf,
                                 bf16* __restrict__ qb, bf16* __restrict__ kb, bf16* __restrict__ vtb,
                                 float* __restrict__ dout) {
  __shared__ __align__(16) bf16 sA[2][64][32];
  __shared__ __align__(16) bf16 sB[2][128][32];
  int tid = threadIdx.x;
  int lane = tid & 31;
  int wave = tid >> 5;
  int wm = wave & 1, wn = wave >> 1;
  int block_m = blockIdx.y * 64;
  int block_n = blockIdx.x * 128;
  int nk = K / 32;

  auto issue_tile = [&](int t, int bufi) {
    int k0 = t * 32;
    unsigned ldsA0 = (unsigned)(unsigned long long)&sA[bufi][0][0];
    unsigned ldsB0 = (unsigned)(unsigned long long)&sB[bufi][0][0];
#pragma unroll
    for (int j = 0; j < 2; ++j) {           /* A tile: 64 rows x 64B = 256 x 16B */
      int id = tid + 128 * j;
      int row = id >> 2, ch = (id & 3) * 8;
      async_load16(ldsA0 + id * 16, A + (size_t)(block_m + row) * K + k0 + ch);
    }
#pragma unroll
    for (int j = 0; j < 4; ++j) {           /* B tile: 128 rows x 64B = 512 x 16B */
      int id = tid + 128 * j;
      int row = id >> 2, ch = (id & 3) * 8;
      async_load16(ldsB0 + id * 16, W + (size_t)(block_n + row) * K + k0 + ch);
    }
  };

  v8f acc[2][4];
#pragma unroll
  for (int i = 0; i < 2; ++i)
#pragma unroll
    for (int t = 0; t < 4; ++t) acc[i][t] = (v8f)0.0f;

  issue_tile(0, 0);
  for (int t = 0; t < nk; ++t) {
    wait_asynccnt0();            /* tile t resident in LDS (this wave's copies) */
    __syncthreads();             /* all waves' copies visible; prev buffer free  */
    if (t + 1 < nk) issue_tile(t + 1, (t + 1) & 1);
    int bi = t & 1;
    v16bf a0 = frag_a_lds(sA[bi], wm * 32, lane);
    v16bf a1 = frag_a_lds(sA[bi], wm * 32 + 16, lane);
#pragma unroll
    for (int tt = 0; tt < 4; ++tt) {
      v16bf bfr = frag_b_lds(sB[bi], wn * 64 + 16 * tt, lane);
      acc[0][tt] = wmma_bf(a0, bfr, acc[0][tt]);
      acc[1][tt] = wmma_bf(a1, bfr, acc[1][tt]);
    }
  }

  int m0 = block_m + wm * 32;
  int n0 = block_n + wn * 64;
  int half = lane >> 4, nl = lane & 15;
#pragma unroll
  for (int i = 0; i < 2; ++i) {
#pragma unroll
    for (int t = 0; t < 4; ++t) {
#pragma unroll
      for (int r = 0; r < 8; ++r) {
        int row = m0 + 16 * i + 8 * half + r;
        int col = n0 + 16 * t + nl;
        float val = acc[i][t][r] + bias[col];
        if constexpr (MODE == 0) {
          int b = row / SPAD, s = row - b * SPAD;
          int part = col / DMODEL, j = col - part * DMODEL;
          int h = j >> 6, dm = j & 63;
          size_t bh = (size_t)(b * NHEADS + h);
          bf16 bv = (bf16)val;
          if (part == 0)      qb[(bh * SPAD + s) * DHEAD + dm] = bv;
          else if (part == 1) kb[(bh * SPAD + s) * DHEAD + dm] = bv;
          else                vtb[(bh * DHEAD + dm) * SPAD + s] = bv;
        } else if constexpr (MODE == 1) {
          size_t idx = (size_t)row * DMODEL + col;
          out_f[idx] = val + resid[idx];
        } else if constexpr (MODE == 2) {
          float g = 0.5f * val * (1.0f + erff(val * 0.70710678f));
          out_bf[(size_t)row * DFF + col] = (bf16)g;
        } else {
          int b = row / SPAD, s = row - b * SPAD;
          if (s < SEQ) {
            float yv = val + resid[(size_t)row * DMODEL + col];
            if (s == 0)          dout[CTX_OFF + (size_t)b * DMODEL + col] = yv;
            else if (s < GTOK)   dout[REG_OFF + ((size_t)b * 4 + (s - 1)) * DMODEL + col] = yv;
            else                 dout[(size_t)b * DMODEL * NPIX + (size_t)col * NPIX + (s - GTOK)] = yv;
          }
        }
      }
    }
  }
}

/* ---- flash attention: 1 wave per (b,h,16-query tile), online softmax ---- */
__global__ void attention_kernel(const bf16* __restrict__ q, const bf16* __restrict__ k,
                                 const bf16* __restrict__ vt, bf16* __restrict__ attn) {
  int lane = threadIdx.x;
  int blk = blockIdx.x;
  int qt = blk % (SPAD / 16);
  int bh = blk / (SPAD / 16);           /* b*12 + h */
  const bf16* qbase = q + ((size_t)bh * SPAD + qt * 16) * DHEAD;
  const bf16* kbase = k + (size_t)bh * SPAD * DHEAD;
  const bf16* vbase = vt + (size_t)bh * DHEAD * SPAD;

  v16bf a0 = load_frag_a(qbase, DHEAD, lane);
  v16bf a1 = load_frag_a(qbase + 32, DHEAD, lane);
  v8f o[4]; float mrun[8], lrun[8];
#pragma unroll
  for (int t = 0; t < 4; ++t) o[t] = (v8f)0.0f;
#pragma unroll
  for (int r = 0; r < 8; ++r) { mrun[r] = -INFINITY; lrun[r] = 0.0f; }

  int half = lane >> 4, nl = lane & 15;
  __shared__ __align__(16) bf16 lds_p[16][32];
  const float scale = 0.125f;  /* 1/sqrt(64) */

  for (int pr = 0; pr < SPAD / 32; ++pr) {
    int j0 = pr * 32;
    v8f s0 = (v8f)0.0f, s1 = (v8f)0.0f;
    s0 = wmma_bf(a0, load_frag_b(kbase + (size_t)j0 * DHEAD, DHEAD, lane), s0);
    s0 = wmma_bf(a1, load_frag_b(kbase + (size_t)j0 * DHEAD + 32, DHEAD, lane), s0);
    s1 = wmma_bf(a0, load_frag_b(kbase + (size_t)(j0 + 16) * DHEAD, DHEAD, lane), s1);
    s1 = wmma_bf(a1, load_frag_b(kbase + (size_t)(j0 + 16) * DHEAD + 32, DHEAD, lane), s1);

    int jc0 = j0 + nl, jc1 = j0 + 16 + nl;
    float p0[8], p1[8];
#pragma unroll
    for (int r = 0; r < 8; ++r) {
      int m = r + 8 * half;
      int qi = qt * 16 + m;
      bool feat = (qi >= GTOK);
      float x0 = s0[r] * scale, x1 = s1[r] * scale;
      if (jc0 >= SEQ || (feat && jc0 < GTOK)) x0 = -INFINITY;
      if (jc1 >= SEQ || (feat && jc1 < GTOK)) x1 = -INFINITY;
      float tmax = fmaxf(x0, x1);
#pragma unroll
      for (int off = 1; off < 16; off <<= 1) tmax = fmaxf(tmax, __shfl_xor(tmax, off, 16));
      float mnew = fmaxf(mrun[r], tmax);
      float c = expf(mrun[r] - mnew);
      float e0 = expf(x0 - mnew), e1 = expf(x1 - mnew);
      float rs = e0 + e1;
#pragma unroll
      for (int off = 1; off < 16; off <<= 1) rs += __shfl_xor(rs, off, 16);
      lrun[r] = lrun[r] * c + rs;
      mrun[r] = mnew;
      p0[r] = e0; p1[r] = e1;
#pragma unroll
      for (int t = 0; t < 4; ++t) o[t][r] = o[t][r] * c;
    }
    __syncthreads();
#pragma unroll
    for (int r = 0; r < 8; ++r) {
      int m = r + 8 * half;
      lds_p[m][nl] = (bf16)p0[r];
      lds_p[m][nl + 16] = (bf16)p1[r];
    }
    __syncthreads();
    FragU pa;
    {
      int rr = lane & 15, hh = lane >> 4;
      v8bf c0, c1;
#pragma unroll
      for (int i2 = 0; i2 < 8; ++i2) {
        c0[i2] = lds_p[rr][hh * 8 + i2];
        c1[i2] = lds_p[rr][hh * 8 + 16 + i2];
      }
      pa.h[0] = c0; pa.h[1] = c1;
    }
#pragma unroll
    for (int t = 0; t < 4; ++t) {
      v16bf vb = load_frag_b(vbase + (size_t)(16 * t) * SPAD + j0, SPAD, lane);
      o[t] = wmma_bf(pa.v, vb, o[t]);
    }
  }
  int b = bh / NHEADS, h = bh - b * NHEADS;
#pragma unroll
  for (int r = 0; r < 8; ++r) {
    int m = r + 8 * half;
    float invl = 1.0f / lrun[r];
    int srow = qt * 16 + m;
#pragma unroll
    for (int t = 0; t < 4; ++t) {
      int dcol = 16 * t + nl;
      attn[((size_t)(b * SPAD + srow)) * DMODEL + h * DHEAD + dcol] = (bf16)(o[t][r] * invl);
    }
  }
}

static size_t ws_take(size_t& off, size_t bytes) {
  size_t a = off;
  off = (off + bytes + 255) & ~(size_t)255;
  return a;
}

extern "C" void kernel_launch(void* const* d_in, const int* in_sizes, int n_in,
                              void* d_out, int out_size, void* d_ws, size_t ws_size,
                              hipStream_t stream) {
  const float* x     = (const float*)d_in[0];
  const float* ctx   = (const float*)d_in[1];
  const float* reg   = (const float*)d_in[2];
  const float* w_in  = (const float*)d_in[3];
  const float* b_in  = (const float*)d_in[4];
  const float* w_out = (const float*)d_in[5];
  const float* b_out = (const float*)d_in[6];
  const float* ln1s  = (const float*)d_in[7];
  const float* ln1b  = (const float*)d_in[8];
  const float* ln2s  = (const float*)d_in[9];
  const float* ln2b  = (const float*)d_in[10];
  const float* w1    = (const float*)d_in[11];
  const float* b1    = (const float*)d_in[12];
  const float* w2    = (const float*)d_in[13];
  const float* b2    = (const float*)d_in[14];
  float* dout = (float*)d_out;
  char* ws = (char*)d_ws;

  size_t off = 0;
  float* xwt_f   = (float*)(ws + ws_take(off, (size_t)MTOT * DMODEL * 4));
  bf16*  xwt_bf  = (bf16*)(ws + ws_take(off, (size_t)MTOT * DMODEL * 2));
  bf16*  win_bf  = (bf16*)(ws + ws_take(off, (size_t)D3 * DMODEL * 2));
  bf16*  wout_bf = (bf16*)(ws + ws_take(off, (size_t)DMODEL * DMODEL * 2));
  bf16*  w1_bf   = (bf16*)(ws + ws_take(off, (size_t)DFF * DMODEL * 2));
  bf16*  w2_bf   = (bf16*)(ws + ws_take(off, (size_t)DMODEL * DFF * 2));
  bf16*  q_bf    = (bf16*)(ws + ws_take(off, (size_t)BATCH * NHEADS * SPAD * DHEAD * 2));
  bf16*  k_bf    = (bf16*)(ws + ws_take(off, (size_t)BATCH * NHEADS * SPAD * DHEAD * 2));
  bf16*  vt_bf   = (bf16*)(ws + ws_take(off, (size_t)BATCH * NHEADS * SPAD * DHEAD * 2));
  bf16*  attn_bf = (bf16*)(ws + ws_take(off, (size_t)MTOT * DMODEL * 2));
  float* y_f     = (float*)(ws + ws_take(off, (size_t)MTOT * DMODEL * 4));
  bf16*  yln_bf  = (bf16*)(ws + ws_take(off, (size_t)MTOT * DMODEL * 2));
  bf16*  h_bf    = (bf16*)(ws + ws_take(off, (size_t)MTOT * DFF * 2));

  auto cvt = [&](const float* s, bf16* d, int n) {
    cvt_bf16_kernel<<<(n + 255) / 256, 256, 0, stream>>>(s, d, n);
  };
  cvt(w_in,  win_bf,  D3 * DMODEL);
  cvt(w_out, wout_bf, DMODEL * DMODEL);
  cvt(w1,    w1_bf,   DFF * DMODEL);
  cvt(w2,    w2_bf,   DMODEL * DFF);

  prep_ln1_kernel<<<MTOT, 256, 0, stream>>>(x, ctx, reg, ln1s, ln1b, xwt_f, xwt_bf);

  /* QKV: (8448 x 2304 x 768) */
  gemm_wmma_kernel<0><<<dim3(D3 / 128, MTOT / 64), 128, 0, stream>>>(
      xwt_bf, win_bf, b_in, DMODEL, D3,
      nullptr, nullptr, nullptr, q_bf, k_bf, vt_bf, nullptr);

  attention_kernel<<<BATCH * NHEADS * (SPAD / 16), 32, 0, stream>>>(q_bf, k_bf, vt_bf, attn_bf);

  /* out-proj + residual(xwt) -> y */
  gemm_wmma_kernel<1><<<dim3(DMODEL / 128, MTOT / 64), 128, 0, stream>>>(
      attn_bf, wout_bf, b_out, DMODEL, DMODEL,
      xwt_f, y_f, nullptr, nullptr, nullptr, nullptr, nullptr);

  ln2_kernel<<<MTOT, 256, 0, stream>>>(y_f, ln2s, ln2b, yln_bf);

  /* FFN1 + GELU */
  gemm_wmma_kernel<2><<<dim3(DFF / 128, MTOT / 64), 128, 0, stream>>>(
      yln_bf, w1_bf, b1, DMODEL, DFF,
      nullptr, nullptr, h_bf, nullptr, nullptr, nullptr, nullptr);

  /* FFN2 + residual(y) -> scattered outputs */
  gemm_wmma_kernel<3><<<dim3(DMODEL / 128, MTOT / 64), 128, 0, stream>>>(
      h_bf, w2_bf, b2, DFF, DMODEL,
      y_f, nullptr, nullptr, nullptr, nullptr, nullptr, dout);

  (void)in_sizes; (void)n_in; (void)out_size; (void)ws_size;
}
